// OrbitGNN_19447611916654
// MI455X (gfx1250) — matrix operand order repaired and checked by reading, not compiled, not run on hardware
//
#include <hip/hip_runtime.h>
#include <hip/hip_bf16.h>

#define GN 100000
#define GE 1600000
#define GETOT (GE + GN)

typedef __attribute__((ext_vector_type(16))) _Float16 v16h;
typedef __attribute__((ext_vector_type(8)))  float    v8f;

// ---------- order-preserving float <-> uint for atomic max ----------
__device__ __forceinline__ unsigned fenc(float f) {
  unsigned b = __float_as_uint(f);
  return b ^ ((unsigned)((int)b >> 31) | 0x80000000u);
}
__device__ __forceinline__ float fdec(unsigned u) {
  unsigned b = (u & 0x80000000u) ? (u ^ 0x80000000u) : ~u;
  return __uint_as_float(b);
}
#define ENC_NEG_INF 0x007FFFFFu  // fenc(-inf)

// ---------- repack B[K,Nn] (f32) into f16 WMMA B-fragment order ----------
// Bp layout: [kTile][nTile][lane 0..31][hi 0..15] halfs; k = kt*32 + (lane>>4)*8 + (hi&7) + (hi>>3)*16,
// n = nt*16 + (lane&15). One thread per packed half.
__global__ void pack_b_f16(const float* __restrict__ B, _Float16* __restrict__ Bp,
                           int K, int Nn) {
  int i = blockIdx.x * blockDim.x + threadIdx.x;
  int total = (K >> 5) * (Nn >> 4) * 512;
  if (i >= total) return;
  int hi   = i & 15;
  int lane = (i >> 4) & 31;
  int tile = i >> 9;
  int nTiles = Nn >> 4;
  int kt = tile / nTiles, nt = tile % nTiles;
  int k = kt * 32 + ((lane >> 4) << 3) + (hi & 7) + ((hi >> 3) << 4);
  int n = nt * 16 + (lane & 15);
  Bp[i] = (_Float16)B[(size_t)k * Nn + n];
}

// ---------- WMMA GEMM: C[M,Nn] = A[M,K] @ B (packed f16), K % 32 == 0 ----------
// one wave per 16x16 tile; blockDim.x/32 waves = Nn/16 column tiles; gridDim.x = M/16
__global__ void gemm_wmma(const float* __restrict__ A, const _Float16* __restrict__ Bp,
                          const float* __restrict__ bias, float* __restrict__ C,
                          int K, int Nn, int do_relu) {
  const int lane   = threadIdx.x & 31;
  const int laneHi = lane >> 4;
  const int lane15 = lane & 15;
  const int m0     = blockIdx.x * 16;
  const int ntile  = threadIdx.x >> 5;
  const int nTilesB = Nn >> 4;
  const int kTiles  = K >> 5;

  const float* arow = A + (size_t)(m0 + lane15) * K + (laneHi << 3);

  v8f acc = {};
  for (int kt = 0; kt < kTiles; ++kt) {
    const float* ar = arow + kt * 32;
    float4 f0 = *(const float4*)(ar);        // K = base+0..3
    float4 f1 = *(const float4*)(ar + 4);    // K = base+4..7
    float4 f2 = *(const float4*)(ar + 16);   // K = base+16..19
    float4 f3 = *(const float4*)(ar + 20);   // K = base+20..23
    v16h a;
    a[0]  = (_Float16)f0.x; a[1]  = (_Float16)f0.y; a[2]  = (_Float16)f0.z; a[3]  = (_Float16)f0.w;
    a[4]  = (_Float16)f1.x; a[5]  = (_Float16)f1.y; a[6]  = (_Float16)f1.z; a[7]  = (_Float16)f1.w;
    a[8]  = (_Float16)f2.x; a[9]  = (_Float16)f2.y; a[10] = (_Float16)f2.z; a[11] = (_Float16)f2.w;
    a[12] = (_Float16)f3.x; a[13] = (_Float16)f3.y; a[14] = (_Float16)f3.z; a[15] = (_Float16)f3.w;

    v16h b = *(const v16h*)(Bp + ((size_t)(kt * nTilesB + ntile) * 32 + lane) * 16);

    acc = __builtin_amdgcn_wmma_f32_16x16x32_f16(false, a, false, b, (short)0, acc,
                                                 false, false);
  }
#pragma unroll
  for (int r = 0; r < 8; ++r) {
    int m = m0 + r + (laneHi << 3);
    int n = ntile * 16 + lane15;
    float v = acc[r];
    if (bias) v += bias[n];
    if (do_relu) v = fmaxf(v, 0.0f);
    C[(size_t)m * Nn + n] = v;
  }
}

// ---------- layer-1 GEMM (K=7, tiny): h1[N,128] = x[N,7] @ W1[7,128] ----------
__global__ void gemm_in7(const float* __restrict__ x, const float* __restrict__ W1,
                         float* __restrict__ h1, int n_nodes) {
  int i = blockIdx.x * blockDim.x + threadIdx.x;
  if (i >= n_nodes * 128) return;
  int n = i >> 7, c = i & 127;
  float acc = 0.0f;
#pragma unroll
  for (int j = 0; j < 7; ++j) acc += x[(size_t)n * 7 + j] * W1[j * 128 + c];
  h1[i] = acc;
}

// ---------- per-node attention scores: es/ed[n,h] = sum_c h[n,h,c]*a[h,c] ----------
__global__ void node_scores(const float* __restrict__ h, const float* __restrict__ a_src,
                            const float* __restrict__ a_dst, float* __restrict__ es,
                            float* __restrict__ ed, int n_nodes) {
  int wave = (int)((blockIdx.x * blockDim.x + threadIdx.x) >> 5);
  int lane = threadIdx.x & 31;
  if (wave >= n_nodes) return;
  int head = lane >> 4;
  int coff = head * 64 + (lane & 15) * 4;
  const float4 hv  = *(const float4*)(h + (size_t)wave * 128 + coff);
  const float4 asv = *(const float4*)(a_src + coff);
  const float4 adv = *(const float4*)(a_dst + coff);
  float es_l = hv.x * asv.x + hv.y * asv.y + hv.z * asv.z + hv.w * asv.w;
  float ed_l = hv.x * adv.x + hv.y * adv.y + hv.z * adv.z + hv.w * adv.w;
#pragma unroll
  for (int m = 8; m >= 1; m >>= 1) {
    es_l += __shfl_xor(es_l, m, 16);
    ed_l += __shfl_xor(ed_l, m, 16);
  }
  if ((lane & 15) == 0) {
    es[wave * 2 + head] = es_l;
    ed[wave * 2 + head] = ed_l;
  }
}

// ---------- init: segment-max keys, segment sums, aggregation buffer ----------
__global__ void init_buf(unsigned* __restrict__ mu, float* __restrict__ ss,
                         float* __restrict__ agg, int n2, int nagg) {
  int i = blockIdx.x * blockDim.x + threadIdx.x;
  if (i < nagg) agg[i] = 0.0f;
  if (i < n2) { mu[i] = ENC_NEG_INF; ss[i] = 0.0f; }
}

// ---------- edge pass 1: e = leaky_relu(es[src]+ed[dst]); atomic segment max ----------
__global__ void edge_logits(const int* __restrict__ ei, const float* __restrict__ es,
                            const float* __restrict__ ed, float* __restrict__ eb,
                            unsigned* __restrict__ mu, int Eo, int Etot) {
  int e = blockIdx.x * blockDim.x + threadIdx.x;
  if (e >= Etot) return;
  int s, d;
  if (e < Eo) { s = ei[e]; d = ei[Eo + e]; } else { s = d = e - Eo; }
#pragma unroll
  for (int h = 0; h < 2; ++h) {
    float v = es[s * 2 + h] + ed[d * 2 + h];
    v = (v >= 0.0f) ? v : 0.2f * v;   // SLOPE = 0.2
    eb[(size_t)e * 2 + h] = v;
    atomicMax(&mu[d * 2 + h], fenc(v));
  }
}

// ---------- edge pass 2: p = exp(e - m[dst]); atomic segment sum ----------
__global__ void edge_exp(const int* __restrict__ ei, float* __restrict__ eb,
                         const unsigned* __restrict__ mu, float* __restrict__ ss,
                         int Eo, int Etot) {
  int e = blockIdx.x * blockDim.x + threadIdx.x;
  if (e >= Etot) return;
  int d = (e < Eo) ? ei[Eo + e] : (e - Eo);
#pragma unroll
  for (int h = 0; h < 2; ++h) {
    float p = __expf(eb[(size_t)e * 2 + h] - fdec(mu[d * 2 + h]));
    eb[(size_t)e * 2 + h] = p;
    atomicAdd(&ss[d * 2 + h], p);
  }
}

// ---------- edge pass 3: agg[dst] += alpha * h[src]  (wave per edge, 4 ch/lane) ----------
__global__ void edge_agg(const int* __restrict__ ei, const float* __restrict__ eb,
                         const float* __restrict__ ss, const float* __restrict__ h,
                         float* __restrict__ agg, int Eo, int Etot) {
  int e = (int)((blockIdx.x * blockDim.x + threadIdx.x) >> 5);
  int lane = threadIdx.x & 31;
  if (e >= Etot) return;
  int s, d;
  if (e < Eo) { s = ei[e]; d = ei[Eo + e]; } else { s = d = e - Eo; }
  int head = lane >> 4;
  float alpha = eb[(size_t)e * 2 + head] / ss[d * 2 + head];
  const float4 hv = *(const float4*)(h + (size_t)s * 128 + lane * 4);
  float* o = agg + (size_t)d * 128 + lane * 4;
  atomicAdd(o + 0, hv.x * alpha);
  atomicAdd(o + 1, hv.y * alpha);
  atomicAdd(o + 2, hv.z * alpha);
  atomicAdd(o + 3, hv.w * alpha);
}

// ---------- layer-1 epilogue: out = relu(agg + b1) ----------
__global__ void bias_relu(float* __restrict__ agg, const float* __restrict__ b, int total) {
  int i = blockIdx.x * blockDim.x + threadIdx.x;
  if (i >= total) return;
  float v = agg[i] + b[i & 127];
  agg[i] = (v > 0.0f) ? v : 0.0f;
}

// ---------- layer-2 epilogue: mean over heads + b2 ----------
__global__ void head_mean(const float* __restrict__ agg, const float* __restrict__ b2,
                          float* __restrict__ out2, int n_nodes) {
  int i = blockIdx.x * blockDim.x + threadIdx.x;
  if (i >= n_nodes * 64) return;
  int n = i >> 6, c = i & 63;
  out2[i] = 0.5f * (agg[(size_t)n * 128 + c] + agg[(size_t)n * 128 + 64 + c]) + b2[c];
}

// ---------- final tiny FC: y[N,4] = t @ Wc2 + bc2 ----------
__global__ void final_fc(const float* __restrict__ t, const float* __restrict__ Wc2,
                         const float* __restrict__ bc2, float* __restrict__ y, int n_nodes) {
  int i = blockIdx.x * blockDim.x + threadIdx.x;
  if (i >= n_nodes * 4) return;
  int n = i >> 2, k = i & 3;
  float acc = bc2[k];
#pragma unroll
  for (int j = 0; j < 64; ++j) acc += t[(size_t)n * 64 + j] * Wc2[j * 4 + k];
  y[i] = acc;
}

extern "C" void kernel_launch(void* const* d_in, const int* in_sizes, int n_in,
                              void* d_out, int out_size, void* d_ws, size_t ws_size,
                              hipStream_t stream) {
  const float* x      = (const float*)d_in[0];
  const int*   ei     = (const int*)  d_in[1];
  const float* W1     = (const float*)d_in[2];
  const float* a_src1 = (const float*)d_in[3];
  const float* a_dst1 = (const float*)d_in[4];
  const float* b1     = (const float*)d_in[5];
  const float* W2     = (const float*)d_in[6];
  const float* a_src2 = (const float*)d_in[7];
  const float* a_dst2 = (const float*)d_in[8];
  const float* b2     = (const float*)d_in[9];
  const float* Wc1    = (const float*)d_in[10];
  const float* bc1    = (const float*)d_in[11];
  const float* Wc2    = (const float*)d_in[12];
  const float* bc2    = (const float*)d_in[13];
  float* y = (float*)d_out;

  // workspace layout (floats)
  float* ws = (float*)d_ws;
  float*    A  = ws;                          // [N,128] h1 / h2 / (out2,t)
  float*    Bb = A  + (size_t)GN * 128;       // [N,128] agg / out1 / agg2
  float*    Cc = Bb + (size_t)GN * 128;       // [Etot,2] edge logits -> p
  float*    ES = Cc + (size_t)GETOT * 2;      // [N,2]
  float*    ED = ES + (size_t)GN * 2;         // [N,2]
  unsigned* MU = (unsigned*)(ED + (size_t)GN * 2); // [N,2]
  float*    SS = (float*)MU + (size_t)GN * 2; // [N,2]
  _Float16* PB2 = (_Float16*)(SS + (size_t)GN * 2); // packed W2: 4*8*512 halfs
  _Float16* PB3 = PB2 + 4 * 8 * 512;                // packed Wc1: 2*4*512 halfs
  float*    OUT2 = A;                         // [N,64] (h2 dead by then)
  float*    T    = A + (size_t)GN * 64;       // [N,64]

  auto cdiv = [](int a, int b) { return (a + b - 1) / b; };
  const int mTiles = GN / 16;        // 6250 (N divisible by 16)
  const int gEdge  = cdiv(GETOT, 256);
  const int gEdgeW = GETOT / 8;      // wave per edge, 8 waves/block (Etot % 8 == 0)
  const int gNodeW = cdiv(GN, 8);    // wave per node, 8 waves/block
  const int gN128  = cdiv(GN * 128, 256);
  const int gN64   = cdiv(GN * 64, 256);
  const int gN4    = cdiv(GN * 4, 256);

  // ===== pack weights for WMMA (tiny, once per call) =====
  pack_b_f16<<<cdiv(4 * 8 * 512, 256), 256, 0, stream>>>(W2, PB2, 128, 128);
  pack_b_f16<<<cdiv(2 * 4 * 512, 256), 256, 0, stream>>>(Wc1, PB3, 64, 64);

  // ===== layer 1 =====
  gemm_in7<<<gN128, 256, 0, stream>>>(x, W1, A, GN);                             // h1
  node_scores<<<gNodeW, 256, 0, stream>>>(A, a_src1, a_dst1, ES, ED, GN);
  init_buf<<<gN128, 256, 0, stream>>>(MU, SS, Bb, GN * 2, GN * 128);
  edge_logits<<<gEdge, 256, 0, stream>>>(ei, ES, ED, Cc, MU, GE, GETOT);
  edge_exp<<<gEdge, 256, 0, stream>>>(ei, Cc, MU, SS, GE, GETOT);
  edge_agg<<<gEdgeW, 256, 0, stream>>>(ei, Cc, SS, A, Bb, GE, GETOT);
  bias_relu<<<gN128, 256, 0, stream>>>(Bb, b1, GN * 128);                        // out1

  // ===== layer 2 =====
  gemm_wmma<<<mTiles, 256, 0, stream>>>(Bb, PB2, nullptr, A, 128, 128, 0);       // h2
  node_scores<<<gNodeW, 256, 0, stream>>>(A, a_src2, a_dst2, ES, ED, GN);
  init_buf<<<gN128, 256, 0, stream>>>(MU, SS, Bb, GN * 2, GN * 128);
  edge_logits<<<gEdge, 256, 0, stream>>>(ei, ES, ED, Cc, MU, GE, GETOT);
  edge_exp<<<gEdge, 256, 0, stream>>>(ei, Cc, MU, SS, GE, GETOT);
  edge_agg<<<gEdgeW, 256, 0, stream>>>(ei, Cc, SS, A, Bb, GE, GETOT);
  head_mean<<<gN64, 256, 0, stream>>>(Bb, b2, OUT2, GN);                         // out2

  // ===== classifier =====
  gemm_wmma<<<mTiles, 128, 0, stream>>>(OUT2, PB3, bc1, T, 64, 64, 1);           // relu(out2@Wc1+bc1)
  final_fc<<<gN4, 256, 0, stream>>>(T, Wc2, bc2, y, GN);
}
